// Attention_1443109012086
// MI455X (gfx1250) — matrix operand order repaired and checked by reading, not compiled, not run on hardware
//
#include <hip/hip_runtime.h>
#include <hip/hip_bf16.h>
#include <cstdint>

// ---------------------------------------------------------------------------
// Problem constants (from reference)
// ---------------------------------------------------------------------------
static constexpr int Bb  = 2;
static constexpr int Ss  = 2048;
static constexpr int Dd  = 4096;
static constexpr int Hh  = 32;
static constexpr int KVH = 8;
static constexpr int HD  = 128;
static constexpr int SW  = 4096;
static constexpr int KV_REP = Hh / KVH;           // 4
#define ATT_SCALE 0.08838834764831845f            // 128^-0.5
#define NEG_INF_MASK (-1000000000.0f)

typedef __attribute__((ext_vector_type(16))) __bf16 bf16x16;
typedef __attribute__((ext_vector_type(8)))  __bf16 bf16x8;
typedef __attribute__((ext_vector_type(8)))  float  f32x8;

__device__ __forceinline__ bf16x16 cat8(bf16x8 lo, bf16x8 hi) {
  return __builtin_shufflevector(lo, hi, 0,1,2,3,4,5,6,7,8,9,10,11,12,13,14,15);
}

// Async L2->LDS 16-byte copy (CDNA5 GLOBAL_LOAD_ASYNC_TO_LDS_B128, ASYNCcnt).
// GVS addressing: 64-bit SGPR base + 32-bit per-lane byte offset.
// LDS destination = low 32 bits of the generic pointer (LDS aperture truncation).
__device__ __forceinline__ void async_copy16(const void* gbase, uint32_t byte_off,
                                             void* lds_ptr) {
  uint32_t lds_addr = (uint32_t)(uintptr_t)lds_ptr;
  asm volatile("global_load_async_to_lds_b128 %0, %1, %2"
               :: "v"(lds_addr), "v"(byte_off), "s"(gbase)
               : "memory");
}
__device__ __forceinline__ void async_wait0() {
  asm volatile("s_wait_asynccnt 0x0" ::: "memory");
}

// ---------------------------------------------------------------------------
// 1) x (f32) -> bf16, same layout
// ---------------------------------------------------------------------------
__global__ __launch_bounds__(256) void convert_f32_bf16(const float* __restrict__ src,
                                                        __bf16* __restrict__ dst, size_t n) {
  size_t i = (size_t)blockIdx.x * 256 + threadIdx.x;
  if (i < n) dst[i] = (__bf16)src[i];
}

// ---------------------------------------------------------------------------
// 2) W (K x N, f32, row-major) -> WT (N x K, bf16) via LDS tile transpose
// ---------------------------------------------------------------------------
__global__ __launch_bounds__(256) void transpose_convert(const float* __restrict__ W,
                                                         __bf16* __restrict__ WT,
                                                         int K, int N) {
  __shared__ float tile[32][33];
  int tx = threadIdx.x & 31, ty = threadIdx.x >> 5;        // 32 x 8
  int n0 = blockIdx.x * 32, k0 = blockIdx.y * 32;
  for (int r = 0; r < 4; ++r) {
    int k = k0 + ty + r * 8;
    tile[ty + r * 8][tx] = W[(size_t)k * N + n0 + tx];
  }
  __syncthreads();
  for (int r = 0; r < 4; ++r) {
    int n = n0 + ty + r * 8;
    WT[(size_t)n * K + k0 + tx] = (__bf16)tile[tx][ty + r * 8];
  }
}

// ---------------------------------------------------------------------------
// 3) bf16 GEMM: C(f32, MxN) = A(bf16, MxK) * BT(bf16, NxK)^T
//    128x128 block tile, BK=64, 8 waves, wave tile 32x64 (2x4 WMMA frags).
//    Double-buffered LDS staging via async global->LDS copies (ASYNCcnt).
// ---------------------------------------------------------------------------
#define GBM 128
#define GBN 128
#define GBK 64
#define KPAD 72   // 64 + 8 bf16 pad -> 144B row stride (16B aligned, conflict-skewed)

__global__ __launch_bounds__(256) void gemm_bf16(const __bf16* __restrict__ A,
                                                 const __bf16* __restrict__ BT,
                                                 float* __restrict__ C,
                                                 int M, int N, int K) {
  __shared__ __align__(16) __bf16 As[2][GBM * KPAD];
  __shared__ __align__(16) __bf16 Bs[2][GBN * KPAD];

  const int tid  = threadIdx.x;
  const int lane = tid & 31;
  const int wid  = tid >> 5;
  const int h4   = lane >> 4;      // which 16-lane half
  const int l16  = lane & 15;
  const int waveM = wid & 3;       // 4 waves along M
  const int waveN = wid >> 2;      // 2 waves along N
  const int m0 = blockIdx.x * GBM;
  const int n0 = blockIdx.y * GBN;

  f32x8 acc[2][4] = {};

  const int cid  = tid & 7;        // 16B chunk within 64-elem row
  const int rrow = tid >> 3;       // 32 rows per staging pass

  // issue async copies for one 128x64 tile pair into buffer `buf`
  auto stage = [&](int buf, int k0) {
    for (int it = 0; it < 4; ++it) {
      int r = rrow + it * 32;
      async_copy16(A, (uint32_t)(((size_t)(m0 + r) * K + k0 + cid * 8) * 2),
                   &As[buf][r * KPAD + cid * 8]);
      async_copy16(BT, (uint32_t)(((size_t)(n0 + r) * K + k0 + cid * 8) * 2),
                   &Bs[buf][r * KPAD + cid * 8]);
    }
  };

  stage(0, 0);

  int buf = 0;
  for (int k0 = 0; k0 < K; k0 += GBK, buf ^= 1) {
    async_wait0();        // my chunks of tile k0 have landed in LDS
    __syncthreads();      // everyone's chunks landed; prev buffer fully consumed

    if (k0 + GBK < K) stage(buf ^ 1, k0 + GBK);   // prefetch next tile

    const __bf16* Abuf = &As[buf][0];
    const __bf16* Bbuf = &Bs[buf][0];
    for (int kk = 0; kk < GBK; kk += 32) {
      // A frags: lane half h4 holds contiguous K = 16h..16h+15
      bf16x16 af[2];
      for (int i = 0; i < 2; ++i) {
        const __bf16* p = &Abuf[(waveM * 32 + i * 16 + l16) * KPAD + kk];
        bf16x8 lo = *(const bf16x8*)(p + 16 * h4);
        bf16x8 hi = *(const bf16x8*)(p + 16 * h4 + 8);
        af[i] = cat8(lo, hi);
      }
      // B frags: lane half h4 holds contiguous K = 16h..16h+15 for column l16
      bf16x16 bfr[4];
      for (int j = 0; j < 4; ++j) {
        const __bf16* p = &Bbuf[(waveN * 64 + j * 16 + l16) * KPAD + kk + 16 * h4];
        bf16x8 lo = *(const bf16x8*)(p);
        bf16x8 hi = *(const bf16x8*)(p + 8);
        bfr[j] = cat8(lo, hi);
      }
      for (int i = 0; i < 2; ++i)
        for (int j = 0; j < 4; ++j)
          acc[i][j] = __builtin_amdgcn_wmma_f32_16x16x32_bf16(
              false, af[i], false, bfr[j], (short)0, acc[i][j], false, false);
    }
  }

  for (int i = 0; i < 2; ++i)
    for (int j = 0; j < 4; ++j)
      for (int e = 0; e < 8; ++e) {
        int row = m0 + waveM * 32 + i * 16 + e + 8 * h4;
        int col = n0 + waveN * 64 + j * 16 + l16;
        C[(size_t)row * N + col] = acc[i][j][e];
      }
}

// ---------------------------------------------------------------------------
// 4) RoPE on Q: qbuf f32 (B*S, H*HD) -> Qb bf16 [B][H][S][HD]
// ---------------------------------------------------------------------------
__global__ __launch_bounds__(256) void rope_q_kernel(const float* __restrict__ qbuf,
                                                     const float* __restrict__ cosp,
                                                     const float* __restrict__ sinp,
                                                     __bf16* __restrict__ Qb) {
  size_t idx = (size_t)blockIdx.x * 256 + threadIdx.x;   // B*S*H*(HD/2)
  int i = idx & 63;
  size_t t = idx >> 6;
  int h = t % Hh; t /= Hh;
  int s = t % Ss; int b = (int)(t / Ss);
  const float* src = qbuf + ((size_t)(b * Ss + s) * Hh + h) * HD;
  float x1 = src[2 * i], x2 = src[2 * i + 1];
  float c = cosp[(size_t)s * 64 + i], sn = sinp[(size_t)s * 64 + i];
  float y1 = x1 * c - x2 * sn;
  float y2 = x1 * sn + x2 * c;
  __bf16* dst = Qb + ((size_t)(b * Hh + h) * Ss + s) * HD;
  dst[2 * i]     = (__bf16)y1;
  dst[2 * i + 1] = (__bf16)y2;
}

// ---------------------------------------------------------------------------
// 5) RoPE on K: kbuf f32 (B*S, KVH*HD) -> cache_k out (f32) + Kb bf16 [B][KVH][S][HD]
// ---------------------------------------------------------------------------
__global__ __launch_bounds__(256) void rope_k_kernel(const float* __restrict__ kbuf,
                                                     const float* __restrict__ cosp,
                                                     const float* __restrict__ sinp,
                                                     float* __restrict__ cache_k_out,
                                                     __bf16* __restrict__ Kb) {
  size_t idx = (size_t)blockIdx.x * 256 + threadIdx.x;   // B*S*KVH*(HD/2)
  int i = idx & 63;
  size_t t = idx >> 6;
  int kv = t % KVH; t /= KVH;
  int s = t % Ss; int b = (int)(t / Ss);
  const float* src = kbuf + ((size_t)(b * Ss + s) * KVH + kv) * HD;
  float x1 = src[2 * i], x2 = src[2 * i + 1];
  float c = cosp[(size_t)s * 64 + i], sn = sinp[(size_t)s * 64 + i];
  float y1 = x1 * c - x2 * sn;
  float y2 = x1 * sn + x2 * c;
  float* ck = cache_k_out + ((size_t)(b * SW + s) * KVH + kv) * HD;
  ck[2 * i]     = y1;
  ck[2 * i + 1] = y2;
  __bf16* dst = Kb + ((size_t)(b * KVH + kv) * Ss + s) * HD;
  dst[2 * i]     = (__bf16)y1;
  dst[2 * i + 1] = (__bf16)y2;
}

// ---------------------------------------------------------------------------
// 6) V: vbuf f32 (B*S, KVH*HD) -> cache_v out (f32) + Vt bf16 [B][KVH][HD][S]
// ---------------------------------------------------------------------------
__global__ __launch_bounds__(256) void copy_v_kernel(const float* __restrict__ vbuf,
                                                     float* __restrict__ cache_v_out,
                                                     __bf16* __restrict__ Vt) {
  size_t idx = (size_t)blockIdx.x * 256 + threadIdx.x;   // B*S*KVH*HD
  int hd = idx & 127;
  size_t t = idx >> 7;
  int kv = t % KVH; t /= KVH;
  int s = t % Ss; int b = (int)(t / Ss);
  float v = vbuf[((size_t)(b * Ss + s) * KVH + kv) * HD + hd];
  cache_v_out[((size_t)(b * SW + s) * KVH + kv) * HD + hd] = v;
  Vt[(((size_t)(b * KVH + kv) * HD) + hd) * Ss + s] = (__bf16)v;
}

// ---------------------------------------------------------------------------
// 7) Zero tail rows [S, SW) of both caches
// ---------------------------------------------------------------------------
__global__ __launch_bounds__(256) void zero_tail_kernel(float* __restrict__ cache_k_out,
                                                        float* __restrict__ cache_v_out) {
  size_t idx = (size_t)blockIdx.x * 256 + threadIdx.x;   // B*(SW-S)*KVH*HD
  int hd = idx & 127;
  size_t t = idx >> 7;
  int kv = t % KVH; t /= KVH;
  int s = Ss + (int)(t % (SW - Ss));
  int b = (int)(t / (SW - Ss));
  size_t off = ((size_t)(b * SW + s) * KVH + kv) * HD + hd;
  cache_k_out[off] = 0.0f;
  cache_v_out[off] = 0.0f;
}

// ---------------------------------------------------------------------------
// 8) Flash attention. Block = (b, h, 128 q-rows), 8 waves, each wave owns
//    16 q-rows and streams causal key blocks of 32. All matmuls via WMMA bf16.
//    Writes attn (bf16) in (b*S+s, h*HD+hd) layout for the epilogue GEMM.
// ---------------------------------------------------------------------------
__global__ __launch_bounds__(256) void attn_kernel(const __bf16* __restrict__ Qb,
                                                   const __bf16* __restrict__ Kb,
                                                   const __bf16* __restrict__ Vt,
                                                   __bf16* __restrict__ attn_bf) {
  constexpr int QBLOCKS = Ss / 128;  // 16
  int bid = blockIdx.x;
  int qbIdx = bid % QBLOCKS;
  int tmp = bid / QBLOCKS;
  int h = tmp % Hh;
  int b = tmp / Hh;

  const int lane = threadIdx.x & 31;
  const int wid  = threadIdx.x >> 5;
  const int h4   = lane >> 4;
  const int l16  = lane & 15;
  const int q0   = qbIdx * 128 + wid * 16;
  const int kv   = h / KV_REP;

  __shared__ __align__(16) __bf16 plds[8 * 16 * 40];  // per-wave 16x32 P scratch (pad 40)
  __bf16* pw = &plds[wid * 16 * 40];

  // Q fragments: 4 chunks of K=32 over HD=128
  bf16x16 qf[4];
  {
    const __bf16* qrow = Qb + ((size_t)(b * Hh + h) * Ss + (q0 + l16)) * HD;
    for (int c = 0; c < 4; ++c) {
      bf16x8 lo = *(const bf16x8*)(qrow + c * 32 + 16 * h4);
      bf16x8 hi = *(const bf16x8*)(qrow + c * 32 + 16 * h4 + 8);
      qf[c] = cat8(lo, hi);
    }
  }

  f32x8 o[8] = {};
  float mrow[8], lrow[8];
  for (int e = 0; e < 8; ++e) { mrow[e] = -1e30f; lrow[e] = 0.0f; }

  const __bf16* kbase = Kb + (size_t)(b * KVH + kv) * Ss * HD;
  const __bf16* vbase = Vt + (size_t)(b * KVH + kv) * HD * Ss;

  const int nkb = q0 / 32 + 1;  // causal: keys up to q0+15 inclusive
  for (int kb = 0; kb < nkb; ++kb) {
    const int k0 = kb * 32;

    // ---- scores S = Q K^T : two 16-key C frags, K-dim = HD in 4 chunks ----
    f32x8 s0 = {}, s1 = {};
    for (int t = 0; t < 2; ++t) {
      const __bf16* krow = kbase + (size_t)(k0 + t * 16 + l16) * HD;
      for (int c = 0; c < 4; ++c) {
        bf16x8 lo = *(const bf16x8*)(krow + c * 32 + 16 * h4);
        bf16x8 hi = *(const bf16x8*)(krow + c * 32 + 16 * h4 + 8);
        bf16x16 kf = cat8(lo, hi);
        if (t == 0)
          s0 = __builtin_amdgcn_wmma_f32_16x16x32_bf16(false, qf[c], false, kf,
                                                       (short)0, s0, false, false);
        else
          s1 = __builtin_amdgcn_wmma_f32_16x16x32_bf16(false, qf[c], false, kf,
                                                       (short)0, s1, false, false);
      }
    }

    // ---- online softmax (per q-row: row e+8*h4, cols across 16-lane half) ----
    float alpha[8];
    for (int e = 0; e < 8; ++e) {
      int r  = q0 + e + 8 * h4;
      int c0 = k0 + l16;
      int c1 = k0 + 16 + l16;
      float v0 = s0[e] * ATT_SCALE; if (c0 > r) v0 = NEG_INF_MASK;
      float v1 = s1[e] * ATT_SCALE; if (c1 > r) v1 = NEG_INF_MASK;
      float mx = fmaxf(v0, v1);
      for (int off = 8; off >= 1; off >>= 1) mx = fmaxf(mx, __shfl_xor(mx, off, 32));
      float mn = fmaxf(mrow[e], mx);
      float al = __expf(mrow[e] - mn);
      float p0 = __expf(v0 - mn);
      float p1 = __expf(v1 - mn);
      float sum = p0 + p1;
      for (int off = 8; off >= 1; off >>= 1) sum += __shfl_xor(sum, off, 32);
      lrow[e] = lrow[e] * al + sum;
      mrow[e] = mn;
      alpha[e] = al;
      s0[e] = p0;
      s1[e] = p1;
    }
    for (int j = 0; j < 8; ++j)
      for (int e = 0; e < 8; ++e) o[j][e] *= alpha[e];

    // ---- transpose P through per-wave LDS into an A fragment (16x32) ----
    for (int e = 0; e < 8; ++e) {
      int row = e + 8 * h4;
      pw[row * 40 + l16]      = (__bf16)s0[e];
      pw[row * 40 + 16 + l16] = (__bf16)s1[e];
    }
    asm volatile("s_wait_dscnt 0" ::: "memory");  // DS is in-order per wave; make the
                                                  // cross-lane readback explicit
    bf16x16 pf;
    {
      const __bf16* p = pw + l16 * 40;
      bf16x8 lo = *(const bf16x8*)(p + 16 * h4);
      bf16x8 hi = *(const bf16x8*)(p + 16 * h4 + 8);
      pf = cat8(lo, hi);
    }

    // ---- O += P V : 8 hd-tiles of 16, contraction over 32 keys ----
    for (int j = 0; j < 8; ++j) {
      const __bf16* vrow = vbase + (size_t)(j * 16 + l16) * Ss + k0 + 16 * h4;
      bf16x8 lo = *(const bf16x8*)(vrow);
      bf16x8 hi = *(const bf16x8*)(vrow + 8);
      bf16x16 vf = cat8(lo, hi);
      o[j] = __builtin_amdgcn_wmma_f32_16x16x32_bf16(false, pf, false, vf,
                                                     (short)0, o[j], false, false);
    }
  }

  // ---- normalize and store (bf16, ready for the wo GEMM) ----
  float inv[8];
  for (int e = 0; e < 8; ++e) inv[e] = 1.0f / lrow[e];
  for (int j = 0; j < 8; ++j)
    for (int e = 0; e < 8; ++e) {
      int r   = q0 + e + 8 * h4;
      int col = j * 16 + l16;
      attn_bf[(size_t)(b * Ss + r) * (Hh * HD) + h * HD + col] =
          (__bf16)(o[j][e] * inv[e]);
    }
}

// ---------------------------------------------------------------------------
// Host orchestration
// ---------------------------------------------------------------------------
extern "C" void kernel_launch(void* const* d_in, const int* in_sizes, int n_in,
                              void* d_out, int out_size, void* d_ws, size_t ws_size,
                              hipStream_t stream) {
  (void)in_sizes; (void)n_in; (void)out_size; (void)ws_size;

  const float* x    = (const float*)d_in[0];
  const float* cosp = (const float*)d_in[1];
  const float* sinp = (const float*)d_in[2];
  // d_in[3] positions (arange), d_in[4] mask (causal), d_in[5,6] zero caches: folded in
  const float* wq   = (const float*)d_in[7];
  const float* wk   = (const float*)d_in[8];
  const float* wv   = (const float*)d_in[9];
  const float* wo   = (const float*)d_in[10];

  float* out_proj = (float*)d_out;                                   // B*S*D
  float* cache_k_out = out_proj + (size_t)Bb * Ss * Dd;              // B*SW*KVH*HD
  float* cache_v_out = cache_k_out + (size_t)Bb * SW * KVH * HD;

  const int M = Bb * Ss;            // 4096
  const size_t MD = (size_t)M * Dd; // 16.7M

  char* ws = (char*)d_ws;
  size_t off = 0;
  auto alloc = [&](size_t bytes) -> void* {
    void* p = ws + off;
    off = (off + bytes + 255) & ~(size_t)255;
    return p;
  };

  __bf16* xb   = (__bf16*)alloc(MD * 2);
  __bf16* wqT  = (__bf16*)alloc((size_t)Dd * (Hh * HD) * 2);
  __bf16* wkT  = (__bf16*)alloc((size_t)Dd * (KVH * HD) * 2);
  __bf16* wvT  = (__bf16*)alloc((size_t)Dd * (KVH * HD) * 2);
  __bf16* woT  = (__bf16*)alloc((size_t)(Hh * HD) * Dd * 2);
  float*  qbuf = (float*)alloc((size_t)M * (Hh * HD) * 4);
  float*  kbuf = (float*)alloc((size_t)M * (KVH * HD) * 4);
  float*  vbuf = (float*)alloc((size_t)M * (KVH * HD) * 4);
  __bf16* Qb   = (__bf16*)alloc((size_t)Bb * Hh * Ss * HD * 2);
  __bf16* Kb   = (__bf16*)alloc((size_t)Bb * KVH * Ss * HD * 2);
  __bf16* Vt   = (__bf16*)alloc((size_t)Bb * KVH * HD * Ss * 2);
  __bf16* attn = (__bf16*)alloc(MD * 2);

  // 1) convert activations + weights (transposed to NxK for uniform staging)
  convert_f32_bf16<<<(MD + 255) / 256, 256, 0, stream>>>(x, xb, MD);
  transpose_convert<<<dim3((Hh * HD) / 32, Dd / 32), 256, 0, stream>>>(wq, wqT, Dd, Hh * HD);
  transpose_convert<<<dim3((KVH * HD) / 32, Dd / 32), 256, 0, stream>>>(wk, wkT, Dd, KVH * HD);
  transpose_convert<<<dim3((KVH * HD) / 32, Dd / 32), 256, 0, stream>>>(wv, wvT, Dd, KVH * HD);
  transpose_convert<<<dim3(Dd / 32, (Hh * HD) / 32), 256, 0, stream>>>(wo, woT, Hh * HD, Dd);

  // 2) QKV projections (bf16 WMMA, f32 accumulate, async double-buffered LDS)
  gemm_bf16<<<dim3(M / GBM, (Hh * HD) / GBN), 256, 0, stream>>>(xb, wqT, qbuf, M, Hh * HD, Dd);
  gemm_bf16<<<dim3(M / GBM, (KVH * HD) / GBN), 256, 0, stream>>>(xb, wkT, kbuf, M, KVH * HD, Dd);
  gemm_bf16<<<dim3(M / GBM, (KVH * HD) / GBN), 256, 0, stream>>>(xb, wvT, vbuf, M, KVH * HD, Dd);

  // 3) RoPE + cache writes + attention-friendly bf16 layouts
  {
    size_t nq = (size_t)Bb * Ss * Hh * (HD / 2);
    rope_q_kernel<<<(nq + 255) / 256, 256, 0, stream>>>(qbuf, cosp, sinp, Qb);
    size_t nk = (size_t)Bb * Ss * KVH * (HD / 2);
    rope_k_kernel<<<(nk + 255) / 256, 256, 0, stream>>>(kbuf, cosp, sinp, cache_k_out, Kb);
    size_t nv = (size_t)Bb * Ss * KVH * HD;
    copy_v_kernel<<<(nv + 255) / 256, 256, 0, stream>>>(vbuf, cache_v_out, Vt);
    size_t nz = (size_t)Bb * (SW - Ss) * KVH * HD;
    zero_tail_kernel<<<(nz + 255) / 256, 256, 0, stream>>>(cache_k_out, cache_v_out);
  }

  // 4) causal flash attention (WMMA bf16)
  attn_kernel<<<Bb * Hh * (Ss / 128), 256, 0, stream>>>(Qb, Kb, Vt, attn);

  // 5) output projection
  gemm_bf16<<<dim3(M / GBM, Dd / GBN), 256, 0, stream>>>(attn, woT, out_proj, M, Dd, Hh * HD);
}
